// E1Cell_2147483648046
// MI455X (gfx1250) — compile-verified, matrix operand order checked by reading, not run on hardware
//
#include <hip/hip_runtime.h>
#include <hip/hip_bf16.h>

// ---------------------------------------------------------------------------
// Gated Elman scan on gfx1250:
//   x_proj = x @ Wx^T + b   (big WMMA bf16 GEMM, fp32 accum)
//   loop t: h = tanh(x_proj_t + h_prev @ Wh^T); y = h * silu(z_t)
// ---------------------------------------------------------------------------

#define T_LEN 512
#define BATCH 32
#define DIM   1280
#define KT_N  (DIM / 32)             // 40 k-steps (even)
#define DD    (DIM * DIM)            // 1,638,400
#define BD    (BATCH * DIM)          // 40,960
#define MROWS (T_LEN * BATCH)        // 16,384
#define XN    (MROWS * DIM)          // 20,971,520

typedef __attribute__((ext_vector_type(16))) __bf16        v16bf;
typedef __attribute__((ext_vector_type(8)))  float         v8f;
typedef __attribute__((ext_vector_type(8)))  unsigned int  v8u;
typedef __attribute__((ext_vector_type(4)))  unsigned int  u32x4;

// fp32 -> bf16, round-to-nearest-even
__device__ __forceinline__ unsigned short f2bf(float f) {
    unsigned int u = __builtin_bit_cast(unsigned int, f);
    unsigned int r = u + 0x7fffu + ((u >> 16) & 1u);
    return (unsigned short)(r >> 16);
}

// 16x16x32 bf16 A/B fragment: p -> element [row=lane&15, k0+(lane>>4)*8].
// Fragment = p[0..7] in v[0..3], p[16..23] in v[4..7].
__device__ __forceinline__ v16bf load_frag_bf16(const unsigned short* p) {
    u32x4 lo = *(const u32x4*)(p);
    u32x4 hi = *(const u32x4*)(p + 16);
    v8u u;
    u[0] = lo[0]; u[1] = lo[1]; u[2] = lo[2]; u[3] = lo[3];
    u[4] = hi[0]; u[5] = hi[1]; u[6] = hi[2]; u[7] = hi[3];
    return __builtin_bit_cast(v16bf, u);
}

__device__ __forceinline__ float silu(float z) {
    return z / (1.0f + __expf(-z));
}

__device__ __forceinline__ v8f wmma_bf16(v16bf a, v16bf b, v8f c) {
    return __builtin_amdgcn_wmma_f32_16x16x32_bf16(
        false, a, false, b, (short)0, c, false, false);
}

// ---------------------------------------------------------------------------
// 1) Convert Wx, Wh, h0 and the whole x_seq to bf16 workspace copies.
//    Pure bandwidth (~145 MB): hoists ALL fp32->bf16 VALU work out of the
//    WMMA inner loops.
// ---------------------------------------------------------------------------
__global__ void prep_kernel(const float* __restrict__ wx,
                            const float* __restrict__ wh,
                            const float* __restrict__ h0,
                            const float* __restrict__ x,
                            unsigned short* __restrict__ wx_bf,
                            unsigned short* __restrict__ wh_bf,
                            unsigned short* __restrict__ h_bf,
                            unsigned short* __restrict__ x_bf) {
    int idx = blockIdx.x * blockDim.x + threadIdx.x;
    if (idx < DD) {
        wx_bf[idx] = f2bf(wx[idx]);
    } else if (idx < 2 * DD) {
        int i = idx - DD;
        wh_bf[i] = f2bf(wh[i]);
    } else if (idx < 2 * DD + BD) {
        int i = idx - 2 * DD;
        h_bf[i] = f2bf(h0[i]);
    } else if (idx < 2 * DD + BD + XN) {
        int i = idx - (2 * DD + BD);
        x_bf[i] = f2bf(__builtin_nontemporal_load(x + i));  // x read once
    }
}

// ---------------------------------------------------------------------------
// 2) x_proj[16384, DIM] = X_bf16 @ Wx^T + b   (fp32 out)
//    One wave = one 16-row M-tile x 4 N-tiles. Two-stage ping-pong pipeline:
//    even-k / odd-k fragments live in distinct values so the rotation is pure
//    register renaming (no v_mov copies).
// ---------------------------------------------------------------------------
__global__ void xproj_kernel(const unsigned short* __restrict__ x_bf,
                             const unsigned short* __restrict__ wx_bf,
                             const float* __restrict__ bias,
                             float* __restrict__ xproj) {
    const int lane = threadIdx.x & 31;
    const int wave = threadIdx.x >> 5;
    const int job  = blockIdx.x * 8 + wave;        // 0 .. 20479
    const int mtile = job / 20;                    // 0 .. 1023
    const int ng    = job % 20;                    // 4 n-tiles each

    const int row  = lane & 15;
    const int koff = (lane >> 4) * 8;

    const unsigned short* ar  = x_bf  + (size_t)(mtile * 16 + row) * DIM + koff;
    const unsigned short* b0r = wx_bf + (size_t)((ng * 4 + 0) * 16 + row) * DIM + koff;
    const unsigned short* b1r = wx_bf + (size_t)((ng * 4 + 1) * 16 + row) * DIM + koff;
    const unsigned short* b2r = wx_bf + (size_t)((ng * 4 + 2) * 16 + row) * DIM + koff;
    const unsigned short* b3r = wx_bf + (size_t)((ng * 4 + 3) * 16 + row) * DIM + koff;

    v8f acc0 = {}, acc1 = {}, acc2 = {}, acc3 = {};

    // Stage A = even k, Stage B = odd k.
    v16bf aA  = load_frag_bf16(ar);
    v16bf b0A = load_frag_bf16(b0r);
    v16bf b1A = load_frag_bf16(b1r);
    v16bf b2A = load_frag_bf16(b2r);
    v16bf b3A = load_frag_bf16(b3r);
    v16bf aB  = load_frag_bf16(ar  + 32);
    v16bf b0B = load_frag_bf16(b0r + 32);
    v16bf b1B = load_frag_bf16(b1r + 32);
    v16bf b2B = load_frag_bf16(b2r + 32);
    v16bf b3B = load_frag_bf16(b3r + 32);

#pragma unroll 1
    for (int i = 0; i < KT_N / 2 - 1; ++i) {       // 19 double-iterations
        const int kA = (2 * i + 2) * 32;
        const int kB = (2 * i + 3) * 32;
        acc0 = wmma_bf16(aA, b0A, acc0);
        acc1 = wmma_bf16(aA, b1A, acc1);
        acc2 = wmma_bf16(aA, b2A, acc2);
        acc3 = wmma_bf16(aA, b3A, acc3);
        aA  = load_frag_bf16(ar  + kA);
        b0A = load_frag_bf16(b0r + kA);
        b1A = load_frag_bf16(b1r + kA);
        b2A = load_frag_bf16(b2r + kA);
        b3A = load_frag_bf16(b3r + kA);
        acc0 = wmma_bf16(aB, b0B, acc0);
        acc1 = wmma_bf16(aB, b1B, acc1);
        acc2 = wmma_bf16(aB, b2B, acc2);
        acc3 = wmma_bf16(aB, b3B, acc3);
        aB  = load_frag_bf16(ar  + kB);
        b0B = load_frag_bf16(b0r + kB);
        b1B = load_frag_bf16(b1r + kB);
        b2B = load_frag_bf16(b2r + kB);
        b3B = load_frag_bf16(b3r + kB);
    }
    // Tail: k = 38 (A) and k = 39 (B).
    acc0 = wmma_bf16(aA, b0A, acc0);
    acc1 = wmma_bf16(aA, b1A, acc1);
    acc2 = wmma_bf16(aA, b2A, acc2);
    acc3 = wmma_bf16(aA, b3A, acc3);
    acc0 = wmma_bf16(aB, b0B, acc0);
    acc1 = wmma_bf16(aB, b1B, acc1);
    acc2 = wmma_bf16(aB, b2B, acc2);
    acc3 = wmma_bf16(aB, b3B, acc3);

    const int mhalf = (lane >> 4) * 8;
    v8f accs[4] = {acc0, acc1, acc2, acc3};
#pragma unroll
    for (int j = 0; j < 4; ++j) {
        const int ncol = (ng * 4 + j) * 16 + (lane & 15);
        const float be = bias[ncol];
#pragma unroll
        for (int r = 0; r < 8; ++r) {
            const int m = mtile * 16 + mhalf + r;
            xproj[(size_t)m * DIM + ncol] = accs[j][r] + be;   // L2-resident
        }
    }
}

// ---------------------------------------------------------------------------
// 3) One recurrence step. One wave = one 16-col N-tile x BOTH M-tiles
//    (Wh fragment reused for 2 WMMAs). 80 waves -> 10 blocks x 256.
//    Same two-stage ping-pong pipeline.
// ---------------------------------------------------------------------------
__global__ void step_kernel(const float* __restrict__ xproj_t,
                            const float* __restrict__ z_t,
                            const unsigned short* __restrict__ h_prev,
                            unsigned short* __restrict__ h_cur,
                            const unsigned short* __restrict__ wh_bf,
                            float* __restrict__ y_t,
                            float* __restrict__ h_last_out) {
    const int lane  = threadIdx.x & 31;
    const int wave  = threadIdx.x >> 5;
    const int ntile = blockIdx.x * 8 + wave;   // 0 .. 79

    const int row  = lane & 15;
    const int koff = (lane >> 4) * 8;

    const unsigned short* a0r = h_prev + (size_t)(row)      * DIM + koff;  // rows 0..15
    const unsigned short* a1r = h_prev + (size_t)(16 + row) * DIM + koff;  // rows 16..31
    const unsigned short* br  = wh_bf  + (size_t)(ntile * 16 + row) * DIM + koff;

    v8f acc0 = {}, acc1 = {};

    v16bf a0A = load_frag_bf16(a0r);
    v16bf a1A = load_frag_bf16(a1r);
    v16bf bA  = load_frag_bf16(br);
    v16bf a0B = load_frag_bf16(a0r + 32);
    v16bf a1B = load_frag_bf16(a1r + 32);
    v16bf bB  = load_frag_bf16(br  + 32);

#pragma unroll 1
    for (int i = 0; i < KT_N / 2 - 1; ++i) {       // 19 double-iterations
        const int kA = (2 * i + 2) * 32;
        const int kB = (2 * i + 3) * 32;
        acc0 = wmma_bf16(a0A, bA, acc0);
        acc1 = wmma_bf16(a1A, bA, acc1);
        a0A = load_frag_bf16(a0r + kA);
        a1A = load_frag_bf16(a1r + kA);
        bA  = load_frag_bf16(br  + kA);
        acc0 = wmma_bf16(a0B, bB, acc0);
        acc1 = wmma_bf16(a1B, bB, acc1);
        a0B = load_frag_bf16(a0r + kB);
        a1B = load_frag_bf16(a1r + kB);
        bB  = load_frag_bf16(br  + kB);
    }
    acc0 = wmma_bf16(a0A, bA, acc0);
    acc1 = wmma_bf16(a1A, bA, acc1);
    acc0 = wmma_bf16(a0B, bB, acc0);
    acc1 = wmma_bf16(a1B, bB, acc1);

    const int ncol  = ntile * 16 + (lane & 15);
    const int mhalf = (lane >> 4) * 8;
    v8f accs[2] = {acc0, acc1};
#pragma unroll
    for (int mt = 0; mt < 2; ++mt) {
#pragma unroll
        for (int r = 0; r < 8; ++r) {
            const int m = mt * 16 + mhalf + r;
            const size_t idx = (size_t)m * DIM + ncol;
            const float hv = tanhf(accs[mt][r] + xproj_t[idx]);
            const float zz = __builtin_nontemporal_load(z_t + idx);
            __builtin_nontemporal_store(hv * silu(zz), y_t + idx);  // touch-once
            h_cur[idx] = f2bf(hv);
            if (h_last_out) h_last_out[idx] = hv;
        }
    }
}

// ---------------------------------------------------------------------------
// Host-side launch: prep -> big GEMM -> 512 step kernels (all on `stream`).
// Workspace layout (bytes):
//   [0,        3276800)   Wx bf16
//   [3276800,  6553600)   Wh bf16
//   [6553600,  6635520)   h ping  (bf16, seeded with h0)
//   [6635520,  6717440)   h pong  (bf16)
//   [6717440,  48660480)  x_seq bf16
//   [48660480, 132546560) x_proj fp32
// ---------------------------------------------------------------------------
extern "C" void kernel_launch(void* const* d_in, const int* in_sizes, int n_in,
                              void* d_out, int out_size, void* d_ws, size_t ws_size,
                              hipStream_t stream) {
    (void)in_sizes; (void)n_in; (void)out_size; (void)ws_size;

    const float* x_seq = (const float*)d_in[0];  // [T,B,D]
    const float* z_seq = (const float*)d_in[1];  // [T,B,D]
    const float* h0    = (const float*)d_in[2];  // [B,D]
    const float* W_x   = (const float*)d_in[3];  // [D,D]
    const float* W_h   = (const float*)d_in[4];  // [D,D]
    const float* bias  = (const float*)d_in[5];  // [D]

    float* out = (float*)d_out;                  // [T,B,D] y then [B,D] h_last

    char* ws = (char*)d_ws;
    unsigned short* wx_bf = (unsigned short*)(ws);
    unsigned short* wh_bf = (unsigned short*)(ws + 3276800u);
    unsigned short* hbuf0 = (unsigned short*)(ws + 6553600u);
    unsigned short* hbuf1 = (unsigned short*)(ws + 6635520u);
    unsigned short* x_bf  = (unsigned short*)(ws + 6717440u);
    float*          xproj = (float*)        (ws + 48660480u);

    // 1) bf16 conversions (re-seeds h ping buffer every call -> replay-safe).
    {
        const long long total = 2LL * DD + BD + XN;   // 24,289,280
        const int blocks = (int)((total + 255) / 256);
        prep_kernel<<<blocks, 256, 0, stream>>>(
            W_x, W_h, h0, x_seq, wx_bf, wh_bf, hbuf0, x_bf);
    }

    // 2) Input projection GEMM: 20480 wave-jobs, 8 waves/block.
    xproj_kernel<<<2560, 256, 0, stream>>>(x_bf, wx_bf, bias, xproj);

    // 3) Sequential scan: 512 small WMMA kernels (graph-captured).
    unsigned short* hb[2] = {hbuf0, hbuf1};
    for (int t = 0; t < T_LEN; ++t) {
        const unsigned short* hp = hb[t & 1];
        unsigned short*       hc = hb[(t + 1) & 1];
        float* hlast = (t == T_LEN - 1) ? (out + (size_t)T_LEN * BD) : nullptr;
        step_kernel<<<10, 256, 0, stream>>>(
            xproj + (size_t)t * BD,
            z_seq + (size_t)t * BD,
            hp, hc, wh_bf,
            out + (size_t)t * BD,
            hlast);
    }
}
